// Encoder_28716151341483
// MI455X (gfx1250) — matrix operand order repaired and checked by reading, not compiled, not run on hardware
//
#include <hip/hip_runtime.h>

typedef unsigned short u16;
typedef unsigned int   u32;
typedef __attribute__((ext_vector_type(16))) __bf16 v16bf;
typedef __attribute__((ext_vector_type(8)))  float  v8f;

struct Bits256 { uint4 lo, hi; };

static __device__ inline v16bf frag_from(uint4 lo, uint4 hi) {
    Bits256 t{lo, hi};
    return __builtin_bit_cast(v16bf, t);
}
static __device__ inline v8f vzero8() {
    v8f z = {0.f,0.f,0.f,0.f,0.f,0.f,0.f,0.f};
    return z;
}
static __device__ inline u16 f2bf(float f) {
    u32 u = __builtin_bit_cast(u32, f);
    u += 0x7fffu + ((u >> 16) & 1u);          // round-to-nearest-even
    return (u16)(u >> 16);
}
static __device__ inline v8f wmma_bf16(v16bf a, v16bf b, v8f c) {
    return __builtin_amdgcn_wmma_f32_16x16x32_bf16(false, a, false, b, (short)0, c, false, false);
}

// ---------------------------------------------------------------- f32 -> bf16 (no transpose)
__global__ __launch_bounds__(256) void cvt_bf16_kernel(const float* __restrict__ in,
                                                       u16* __restrict__ out, int n4) {
    int i = blockIdx.x * 256 + threadIdx.x;
    if (i >= n4) return;
    float4 f = ((const float4*)in)[i];
    uint2 o;
    o.x = (u32)f2bf(f.x) | ((u32)f2bf(f.y) << 16);
    o.y = (u32)f2bf(f.z) | ((u32)f2bf(f.w) << 16);
    ((uint2*)out)[i] = o;
}

// ---------------------------------------------------------------- f32 [K][N] -> bf16 [N][K]
// Tiled 64x64 transpose through LDS; coalesced in, coalesced out.
__global__ __launch_bounds__(256) void cvt_bf16_t_kernel(const float* __restrict__ in,
                                                         u16* __restrict__ out,
                                                         int K, int N) {
    __shared__ u16 tile[64 * 64];
    const int tid = threadIdx.x;
    const int n0 = blockIdx.x * 64;
    const int k0 = blockIdx.y * 64;
    // load 64(k) x 64(n) as float4, convert, store [k][n]
#pragma unroll
    for (int t = 0; t < 4; ++t) {
        int chunk = tid + t * 256;           // 1024 chunks of 4 floats
        int kk = chunk >> 4, n4 = chunk & 15;
        float4 f = *(const float4*)(in + (size_t)(k0 + kk) * N + n0 + n4 * 4);
        uint2 o;
        o.x = (u32)f2bf(f.x) | ((u32)f2bf(f.y) << 16);
        o.y = (u32)f2bf(f.z) | ((u32)f2bf(f.w) << 16);
        ((uint2*)tile)[chunk] = o;
    }
    __syncthreads();
    // write transposed rows [n][k] as uint4
#pragma unroll
    for (int t = 0; t < 2; ++t) {
        int chunk = tid + t * 256;           // 512 chunks of 8 u16
        int nn = chunk >> 3, k8 = chunk & 7;
        u32 w[4];
#pragma unroll
        for (int i = 0; i < 4; ++i) {
            u32 lo = tile[(k8 * 8 + 2 * i)     * 64 + nn];
            u32 hi = tile[(k8 * 8 + 2 * i + 1) * 64 + nn];
            w[i] = lo | (hi << 16);
        }
        uint4 v; v.x = w[0]; v.y = w[1]; v.z = w[2]; v.w = w[3];
        *(uint4*)(out + (size_t)(n0 + nn) * K + k0 + k8 * 8) = v;
    }
}

// ---------------------------------------------------------------- GEMM (bf16 in, f32 acc)
// C[M,N] = scale * (A[M,K] @ B[K,N] + bias[N]); B given PRE-TRANSPOSED as Bt[N][K].
// BM=BN=128, BK=32; 256 threads = 8 waves (4M x 2N); wave tile 32x64.
// Double-buffered LDS + register prefetch: one barrier per K-step.
// ACT: 0=none, 1=exact GELU. OUT_MODE: 0=f32 [M][N], 1=bf16 [M][N], 2=bf16 transposed [N][M].
template <int ACT, int OUT_MODE>
__global__ __launch_bounds__(256) void gemm_bf16_kernel(
        const u16* __restrict__ A, const u16* __restrict__ Bt,
        const float* __restrict__ bias, void* __restrict__ Cout,
        int M, int N, int K, float scale) {
    __shared__ u16 As[2 * 128 * 32];   // [buf][row][k]
    __shared__ u16 Bs[2 * 128 * 32];   // [buf][col][k]

    const int tid  = threadIdx.x;
    const int lane = tid & 31;
    const int wid  = tid >> 5;
    const int wm   = wid & 3;
    const int wn   = wid >> 2;
    const int ln   = lane & 15;
    const int hs   = (lane < 16) ? 0 : 1;

    const int bm = blockIdx.y;
    const int bn = blockIdx.x;
    const int KT = K >> 5;

    v8f acc[2][4];
#pragma unroll
    for (int mi = 0; mi < 2; ++mi)
#pragma unroll
        for (int ni = 0; ni < 4; ++ni) acc[mi][ni] = vzero8();

    uint4 ga[2], gb[2];
    auto load_tile = [&](int kt) {
        const int k0 = kt << 5;
#pragma unroll
        for (int t = 0; t < 2; ++t) {
            int chunk = tid + t * 256;        // 512 chunks of 8 u16
            int row = chunk >> 2, c8 = chunk & 3;
            ga[t] = *(const uint4*)(A  + (size_t)(bm * 128 + row) * K + k0 + c8 * 8);
            gb[t] = *(const uint4*)(Bt + (size_t)(bn * 128 + row) * K + k0 + c8 * 8);
        }
    };
    auto store_tile = [&](int buf) {
#pragma unroll
        for (int t = 0; t < 2; ++t) {
            int chunk = tid + t * 256;
            ((uint4*)(As + buf * 4096))[chunk] = ga[t];
            ((uint4*)(Bs + buf * 4096))[chunk] = gb[t];
        }
    };

    load_tile(0);
    store_tile(0);
    __syncthreads();

    for (int kt = 0; kt < KT; ++kt) {
        const int cur = kt & 1;
        if (kt + 1 < KT) load_tile(kt + 1);   // prefetch next tile into registers

        const u16* as = As + cur * 4096;
        const u16* bs = Bs + cur * 4096;
        v16bf afr[2];
#pragma unroll
        for (int mi = 0; mi < 2; ++mi) {
            const uint4* rp = (const uint4*)(as + (wm * 32 + mi * 16 + ln) * 32);
            afr[mi] = frag_from(rp[hs], rp[2 + hs]);
        }
#pragma unroll
        for (int ni = 0; ni < 4; ++ni) {
            const uint4* bp = (const uint4*)(bs + (wn * 64 + ni * 16 + ln) * 32);
            v16bf bfr = frag_from(bp[hs], bp[2 + hs]);
#pragma unroll
            for (int mi = 0; mi < 2; ++mi)
                acc[mi][ni] = wmma_bf16(afr[mi], bfr, acc[mi][ni]);
        }

        if (kt + 1 < KT) store_tile((kt + 1) & 1);
        __syncthreads();
    }

    // epilogue (branch-free via templates)
#pragma unroll
    for (int mi = 0; mi < 2; ++mi) {
#pragma unroll
        for (int ni = 0; ni < 4; ++ni) {
            int col = bn * 128 + wn * 64 + ni * 16 + ln;
            float bv = bias[col];
#pragma unroll
            for (int r = 0; r < 8; ++r) {
                int row = bm * 128 + wm * 32 + mi * 16 + r + hs * 8;
                float v = (acc[mi][ni][r] + bv) * scale;
                if (ACT == 1) v = 0.5f * v * (1.f + erff(v * 0.70710678118f));
                if (OUT_MODE == 0) ((float*)Cout)[(size_t)row * N + col] = v;
                if (OUT_MODE == 1) ((u16*)Cout)[(size_t)row * N + col] = f2bf(v);
                if (OUT_MODE == 2) ((u16*)Cout)[(size_t)col * M + row] = f2bf(v);
            }
        }
    }
}

// ---------------------------------------------------------------- flash attention
// grid: (S/64 query tiles, 12 heads); block = 128 threads = 4 waves; wave = 16 query rows.
// K given as [S][768] bf16; V given PRE-TRANSPOSED as Vt[768][S] bf16.
// Double-buffered K/V tiles + register prefetch: one barrier per key block.
__global__ __launch_bounds__(128) void attn_kernel(
        const u16* __restrict__ Q, const u16* __restrict__ Kb,
        const u16* __restrict__ Vtg, const int* __restrict__ mask,
        u16* __restrict__ Ctx) {
    constexpr int D = 768, S = 4096;
    __shared__ u16 Ks[2 * 64 * 64];    // [buf][key][d]
    __shared__ u16 Vt[2 * 64 * 64];    // [buf][d][key]
    __shared__ u16 Ps[4 * 16 * 64];    // per-wave P staging [wave*16+row][key]

    const int tid  = threadIdx.x;
    const int lane = tid & 31;
    const int wid  = tid >> 5;
    const int ln   = lane & 15;
    const int hs   = (lane < 16) ? 0 : 1;

    const int h  = blockIdx.y;
    const int ho = h * 64;
    const int q0 = blockIdx.x * 64 + wid * 16;

    // preload Q fragments (two K-steps over d=0..63)
    const u16* qrow = Q + (size_t)(q0 + ln) * D + ho;
    v16bf qf[2];
#pragma unroll
    for (int ks = 0; ks < 2; ++ks) {
        const uint4* qp = (const uint4*)qrow;
        qf[ks] = frag_from(qp[ks * 4 + hs], qp[ks * 4 + 2 + hs]);
    }

    float mrow[8], lrow[8];
    v8f o[4];
#pragma unroll
    for (int r = 0; r < 8; ++r) { mrow[r] = -1e30f; lrow[r] = 0.f; }
#pragma unroll
    for (int ni = 0; ni < 4; ++ni) o[ni] = vzero8();

    uint4 gk[4], gv[4];
    auto load_kv = [&](int kb) {
#pragma unroll
        for (int t = 0; t < 4; ++t) {
            int chunk = tid + t * 128;        // 512 chunks of 8 u16
            int r0 = chunk >> 3, c8 = chunk & 7;
            gk[t] = *(const uint4*)(Kb  + (size_t)(kb * 64 + r0) * D + ho + c8 * 8);
            gv[t] = *(const uint4*)(Vtg + (size_t)(ho + r0) * S + kb * 64 + c8 * 8);
        }
    };
    auto store_kv = [&](int buf) {
#pragma unroll
        for (int t = 0; t < 4; ++t) {
            int chunk = tid + t * 128;
            ((uint4*)(Ks + buf * 4096))[chunk] = gk[t];
            ((uint4*)(Vt + buf * 4096))[chunk] = gv[t];
        }
    };

    load_kv(0);
    store_kv(0);
    __syncthreads();

    for (int kb = 0; kb < 64; ++kb) {
        const int cur = kb & 1;
        if (kb < 63) load_kv(kb + 1);         // prefetch next K/V block

        const u16* ks_ = Ks + cur * 4096;
        const u16* vt_ = Vt + cur * 4096;

        // S = Q @ K^T
        v8f s[4];
#pragma unroll
        for (int ni = 0; ni < 4; ++ni) s[ni] = vzero8();
#pragma unroll
        for (int ks = 0; ks < 2; ++ks) {
#pragma unroll
            for (int ni = 0; ni < 4; ++ni) {
                const uint4* kp = (const uint4*)(ks_ + (ni * 16 + ln) * 64);
                v16bf bf = frag_from(kp[ks * 4 + hs], kp[ks * 4 + 2 + hs]);
                s[ni] = wmma_bf16(qf[ks], bf, s[ni]);
            }
        }
        // scale + mask
#pragma unroll
        for (int ni = 0; ni < 4; ++ni) {
            int key = kb * 64 + ni * 16 + ln;
            int mk = mask[key];
#pragma unroll
            for (int r = 0; r < 8; ++r) {
                float v = s[ni][r] * 0.125f;  // 1/sqrt(64)
                s[ni][r] = mk ? v : -1e30f;
            }
        }
        // row max across fragments, then across the 16-lane half-wave
        float bm8[8];
#pragma unroll
        for (int r = 0; r < 8; ++r)
            bm8[r] = fmaxf(fmaxf(s[0][r], s[1][r]), fmaxf(s[2][r], s[3][r]));
#pragma unroll
        for (int msk = 1; msk <= 8; msk <<= 1)
#pragma unroll
            for (int r = 0; r < 8; ++r)
                bm8[r] = fmaxf(bm8[r], __shfl_xor(bm8[r], msk, 32));
        // online softmax
        float sc8[8], rs[8];
#pragma unroll
        for (int r = 0; r < 8; ++r) {
            float mn = fmaxf(mrow[r], bm8[r]);
            sc8[r] = __expf(mrow[r] - mn);
            mrow[r] = mn;
            rs[r] = 0.f;
        }
#pragma unroll
        for (int ni = 0; ni < 4; ++ni)
#pragma unroll
            for (int r = 0; r < 8; ++r) {
                float p = __expf(s[ni][r] - mrow[r]);
                s[ni][r] = p;
                rs[r] += p;
            }
#pragma unroll
        for (int msk = 1; msk <= 8; msk <<= 1)
#pragma unroll
            for (int r = 0; r < 8; ++r)
                rs[r] += __shfl_xor(rs[r], msk, 32);
#pragma unroll
        for (int r = 0; r < 8; ++r) lrow[r] = lrow[r] * sc8[r] + rs[r];
#pragma unroll
        for (int ni = 0; ni < 4; ++ni)
#pragma unroll
            for (int r = 0; r < 8; ++r) o[ni][r] *= sc8[r];

        // stage P as bf16 (C layout -> A layout), wave-private LDS
#pragma unroll
        for (int ni = 0; ni < 4; ++ni)
#pragma unroll
            for (int r = 0; r < 8; ++r)
                Ps[(wid * 16 + r + hs * 8) * 64 + ni * 16 + ln] = f2bf(s[ni][r]);

        // O += P @ V
        const uint4* prow = (const uint4*)(Ps + (wid * 16 + ln) * 64);
#pragma unroll
        for (int ks = 0; ks < 2; ++ks) {
            v16bf ap = frag_from(prow[ks * 4 + hs], prow[ks * 4 + 2 + hs]);
#pragma unroll
            for (int ni = 0; ni < 4; ++ni) {
                const uint4* vp = (const uint4*)(vt_ + (ni * 16 + ln) * 64);
                v16bf bv = frag_from(vp[ks * 4 + hs], vp[ks * 4 + 2 + hs]);
                o[ni] = wmma_bf16(ap, bv, o[ni]);
            }
        }

        if (kb < 63) store_kv((kb + 1) & 1);
        __syncthreads();
    }

    // normalize and write ctx (bf16) in [S, 768] layout
    float inv8[8];
#pragma unroll
    for (int r = 0; r < 8; ++r) inv8[r] = (lrow[r] > 0.f) ? (1.f / lrow[r]) : 0.f;
#pragma unroll
    for (int ni = 0; ni < 4; ++ni)
#pragma unroll
        for (int r = 0; r < 8; ++r) {
            int row = q0 + r + hs * 8;
            int col = ho + ni * 16 + ln;
            Ctx[(size_t)row * D + col] = f2bf(o[ni][r] * inv8[r]);
        }
}

// ---------------------------------------------------------------- LayerNorm (768 per row)
__global__ __launch_bounds__(256) void ln_kernel(
        const float* __restrict__ A, const float* __restrict__ Res,
        const float* __restrict__ g, const float* __restrict__ b,
        float* __restrict__ Yf, u16* __restrict__ Yb) {
    __shared__ float red[256];
    const int row = blockIdx.x;
    const int tid = threadIdx.x;
    float v[3];
#pragma unroll
    for (int i = 0; i < 3; ++i) {
        int c = tid + i * 256;
        size_t idx = (size_t)row * 768 + c;
        v[i] = A[idx] + (Res ? Res[idx] : 0.f);
    }
    float s = v[0] + v[1] + v[2];
    red[tid] = s; __syncthreads();
    for (int st = 128; st > 0; st >>= 1) {
        if (tid < st) red[tid] += red[tid + st];
        __syncthreads();
    }
    float mu = red[0] * (1.f / 768.f);
    __syncthreads();
    float sq = 0.f;
#pragma unroll
    for (int i = 0; i < 3; ++i) { float d = v[i] - mu; sq += d * d; }
    red[tid] = sq; __syncthreads();
    for (int st = 128; st > 0; st >>= 1) {
        if (tid < st) red[tid] += red[tid + st];
        __syncthreads();
    }
    float rstd = rsqrtf(red[0] * (1.f / 768.f) + 1e-5f);
#pragma unroll
    for (int i = 0; i < 3; ++i) {
        int c = tid + i * 256;
        size_t idx = (size_t)row * 768 + c;
        float y = (v[i] - mu) * rstd * g[c] + b[c];
        if (Yf) Yf[idx] = y;
        if (Yb) Yb[idx] = f2bf(y);
    }
}

// ---------------------------------------------------------------- launcher
extern "C" void kernel_launch(void* const* d_in, const int* in_sizes, int n_in,
                              void* d_out, int out_size, void* d_ws, size_t ws_size,
                              hipStream_t stream) {
    (void)in_sizes; (void)n_in; (void)out_size; (void)ws_size;
    constexpr int S = 4096, D = 768, F = 3072;

    const float* x    = (const float*)d_in[0];
    const int*   mask = (const int*)  d_in[1];
    const float* Wq = (const float*)d_in[2];  const float* bq = (const float*)d_in[3];
    const float* Wk = (const float*)d_in[4];  const float* bk = (const float*)d_in[5];
    const float* Wv = (const float*)d_in[6];  const float* bv = (const float*)d_in[7];
    const float* Wo = (const float*)d_in[8];  const float* bo = (const float*)d_in[9];
    const float* W1 = (const float*)d_in[10]; const float* b1 = (const float*)d_in[11];
    const float* W2 = (const float*)d_in[12]; const float* b2 = (const float*)d_in[13];
    const float* g1 = (const float*)d_in[14]; const float* be1 = (const float*)d_in[15];
    const float* g2 = (const float*)d_in[16]; const float* be2 = (const float*)d_in[17];

    char* p = (char*)d_ws;
    auto alloc = [&](size_t bytes) -> void* {
        void* r = (void*)p;
        p += (bytes + 255) & ~(size_t)255;
        return r;
    };
    u16* xb   = (u16*)alloc((size_t)S * D * 2);
    u16* wqt  = (u16*)alloc((size_t)D * D * 2);   // [N][K] transposed bf16 weights
    u16* wkt  = (u16*)alloc((size_t)D * D * 2);
    u16* wvt  = (u16*)alloc((size_t)D * D * 2);
    u16* wot  = (u16*)alloc((size_t)D * D * 2);
    u16* w1t  = (u16*)alloc((size_t)D * F * 2);
    u16* w2t  = (u16*)alloc((size_t)F * D * 2);
    u16* qb   = (u16*)alloc((size_t)S * D * 2);
    u16* kb   = (u16*)alloc((size_t)S * D * 2);
    u16* vtb  = (u16*)alloc((size_t)S * D * 2);   // V transposed: [768][S]
    u16* ctxb = (u16*)alloc((size_t)S * D * 2);
    float* attnf = (float*)alloc((size_t)S * D * 4);
    float* hf    = (float*)alloc((size_t)S * D * 4);
    u16*   hb    = (u16*)  alloc((size_t)S * D * 2);
    u16*   ff1b  = (u16*)  alloc((size_t)S * F * 2);
    float* ff2f  = (float*)alloc((size_t)S * D * 4);

    // x: plain convert (A operand)
    {
        int n4 = S * D / 4;
        cvt_bf16_kernel<<<(n4 + 255) / 256, 256, 0, stream>>>(x, xb, n4);
    }
    // weights: convert + transpose ([K][N] f32 -> [N][K] bf16)
    cvt_bf16_t_kernel<<<dim3(D / 64, D / 64), 256, 0, stream>>>(Wq, wqt, D, D);
    cvt_bf16_t_kernel<<<dim3(D / 64, D / 64), 256, 0, stream>>>(Wk, wkt, D, D);
    cvt_bf16_t_kernel<<<dim3(D / 64, D / 64), 256, 0, stream>>>(Wv, wvt, D, D);
    cvt_bf16_t_kernel<<<dim3(D / 64, D / 64), 256, 0, stream>>>(Wo, wot, D, D);
    cvt_bf16_t_kernel<<<dim3(F / 64, D / 64), 256, 0, stream>>>(W1, w1t, D, F);
    cvt_bf16_t_kernel<<<dim3(D / 64, F / 64), 256, 0, stream>>>(W2, w2t, F, D);

    // Q, K projections (bf16 [S][D]); V projection written transposed (bf16 [D][S])
    gemm_bf16_kernel<0, 1><<<dim3(D / 128, S / 128), 256, 0, stream>>>(
        xb, wqt, bq, qb, S, D, D, 1.f);
    gemm_bf16_kernel<0, 1><<<dim3(D / 128, S / 128), 256, 0, stream>>>(
        xb, wkt, bk, kb, S, D, D, 1.f);
    gemm_bf16_kernel<0, 2><<<dim3(D / 128, S / 128), 256, 0, stream>>>(
        xb, wvt, bv, vtb, S, D, D, 1.f);

    // flash attention -> ctx (bf16)
    attn_kernel<<<dim3(S / 64, 12), 128, 0, stream>>>(qb, kb, vtb, mask, ctxb);

    // 2*(ctx@Wo + bo): residual attn+attn fused via scale
    gemm_bf16_kernel<0, 0><<<dim3(D / 128, S / 128), 256, 0, stream>>>(
        ctxb, wot, bo, attnf, S, D, D, 2.f);

    // h = LN1(2*attn_out)  (f32 + bf16 copies)
    ln_kernel<<<S, 256, 0, stream>>>(attnf, nullptr, g1, be1, hf, hb);

    // ff1 = gelu(h@W1 + b1) (bf16)
    gemm_bf16_kernel<1, 1><<<dim3(F / 128, S / 128), 256, 0, stream>>>(
        hb, w1t, b1, ff1b, S, F, D, 1.f);

    // ff2 = ff1@W2 + b2 (f32)
    gemm_bf16_kernel<0, 0><<<dim3(D / 128, S / 128), 256, 0, stream>>>(
        ff1b, w2t, b2, ff2f, S, D, F, 1.f);

    // out = LN2(h + ff2)
    ln_kernel<<<S, 256, 0, stream>>>(ff2f, hf, g2, be2, (float*)d_out, nullptr);
}
